// SemanticMemory_39822936769254
// MI455X (gfx1250) — matrix operand order repaired and checked by reading, not compiled, not run on hardware
//
#include <hip/hip_runtime.h>

#define CAP 65536
#define DIM 1024
#define NBATCH 2048
#define TOPK 16
#define MROWS 32                       // rows per wave in the scores kernel
#define SEGS 32
#define KEYS_PER_SEG (CAP / SEGS)      // 2048
#define CHUNK 64
#define NCHUNKS (KEYS_PER_SEG / CHUNK) // 32
#define STRIDE 68                      // LDS row stride in floats (bank-conflict-free)

typedef __bf16 bf16_t;
typedef __attribute__((ext_vector_type(16))) __bf16 v16bf;
typedef __attribute__((ext_vector_type(4)))  __bf16 v4bf;
typedef __attribute__((ext_vector_type(8)))  float  v8f;
typedef __attribute__((ext_vector_type(4)))  float  v4f;
typedef __attribute__((ext_vector_type(4)))  unsigned int v4u;

struct V8U { v4u a, b; }; // 32 bytes, trivially copyable

static __device__ __forceinline__ v16bf frag(v4u lo, v4u hi) {
    V8U t; t.a = lo; t.b = hi;
    return __builtin_bit_cast(v16bf, t);
}

static __device__ __forceinline__ v16bf cvt16(v4f a, v4f b, v4f c, v4f d) {
    v16bf r;
    r[0] = (__bf16)a[0];  r[1] = (__bf16)a[1];  r[2] = (__bf16)a[2];  r[3] = (__bf16)a[3];
    r[4] = (__bf16)b[0];  r[5] = (__bf16)b[1];  r[6] = (__bf16)b[2];  r[7] = (__bf16)b[3];
    r[8] = (__bf16)c[0];  r[9] = (__bf16)c[1];  r[10] = (__bf16)c[2]; r[11] = (__bf16)c[3];
    r[12] = (__bf16)d[0]; r[13] = (__bf16)d[1]; r[14] = (__bf16)d[2]; r[15] = (__bf16)d[3];
    return r;
}

static __device__ __forceinline__ v8f wmma_bf16(v16bf a, v16bf b, v8f c) {
    // D(f32 16x16) = A(16x32 bf16) * B(32x16 bf16) + C
    return __builtin_amdgcn_wmma_f32_16x16x32_bf16(false, a, false, b, (short)0, c,
                                                   false, false);
}

// Sorted-descending insert into a register-resident top-16 list (fully unrolled).
static __device__ __forceinline__ void ins16(float (&ts)[TOPK], int (&ti)[TOPK],
                                             float s, int id) {
    float cs = s; int ci = id;
#pragma unroll
    for (int j = 0; j < TOPK; ++j) {
        bool c = cs > ts[j];
        float os = ts[j]; int oi = ti[j];
        ts[j] = c ? cs : os;  ti[j] = c ? ci : oi;
        cs    = c ? os : cs;  ci    = c ? oi : ci;
    }
}

// ---------------- Kernel 1: keys f32 -> bf16 (makes keys L2-resident: 128MB) -----------
__global__ void k_cvt_keys(const float* __restrict__ kf, bf16_t* __restrict__ kb) {
    size_t i = ((size_t)blockIdx.x * blockDim.x + threadIdx.x) * 4;
    v4f f = *(const v4f*)(kf + i);
    v4bf o;
    o[0] = (__bf16)f[0]; o[1] = (__bf16)f[1]; o[2] = (__bf16)f[2]; o[3] = (__bf16)f[3];
    *(v4bf*)(kb + i) = o;
}

// ---------------- Kernel 2: q = query @ W^T + b, stored as bf16 ------------------------
// One wave computes a 16(M) x 64(N) tile. 2048 waves = 256 blocks x 8 waves.
__global__ void k_compute_q(const float* __restrict__ query, const float* __restrict__ W,
                            const float* __restrict__ bias, bf16_t* __restrict__ qb) {
    int wave = threadIdx.x >> 5, lane = threadIdx.x & 31;
    int gid  = blockIdx.x * 8 + wave;
    int rb   = gid >> 4;            // row block 0..127
    int cb   = (gid & 15) * 64;     // col base
    int half = lane >> 4, lm = lane & 15;
    size_t row = (size_t)rb * 16 + lm;

    v8f z = {0.f,0.f,0.f,0.f,0.f,0.f,0.f,0.f};
    v8f acc[4];
#pragma unroll
    for (int t = 0; t < 4; ++t) acc[t] = z;

    for (int k0 = 0; k0 < DIM; k0 += 32) {
        // A fragment: rows of query, K = k0 + half*16 + {0..15}
        const v4f* ap  = (const v4f*)(query + row * DIM + k0 + half * 16);
        const v4f* ap2 = (const v4f*)(query + row * DIM + k0 + 16 + half * 8);
        v16bf A = cvt16(ap[0], ap[1], ap[2], ap[3]);
#pragma unroll
        for (int nt = 0; nt < 4; ++nt) {
            size_t col = (size_t)cb + nt * 16 + lm;   // W row index (N)
            const v4f* bp = (const v4f*)(W + col * DIM + k0 + half * 16);
            v16bf B = cvt16(bp[0], bp[1], bp[2], bp[3]);
            acc[nt] = wmma_bf16(A, B, acc[nt]);
        }
    }
#pragma unroll
    for (int nt = 0; nt < 4; ++nt) {
        float bj = bias[cb + nt * 16 + lm];
#pragma unroll
        for (int r = 0; r < 8; ++r) {
            int m = r + half * 8;
            float v = acc[nt][r] + bj;
            qb[((size_t)rb * 16 + m) * DIM + cb + nt * 16 + lm] = (bf16_t)v;
        }
    }
}

// ---------------- Kernel 3: fused scores GEMM + per-segment top-16 ---------------------
// One wave: 32 rows x one 2048-key segment, in 64-key chunks.
// 64 row-blocks x 32 segments = 2048 waves = 256 blocks x 8 waves.
template <bool KB>
__global__ void k_scores_topk(const bf16_t* __restrict__ qb, const bf16_t* __restrict__ kb,
                              const float* __restrict__ kf,
                              float* __restrict__ ps, int* __restrict__ pi) {
    __shared__ float lds[8][MROWS * STRIDE];
    int wave = threadIdx.x >> 5, lane = threadIdx.x & 31;
    int gid  = blockIdx.x * 8 + wave;
    int rb   = gid >> 5;            // row block 0..63
    int seg  = gid & 31;            // key segment 0..31
    int half = lane >> 4, lm = lane & 15;
    int row0 = rb * MROWS;
    int key0seg = seg * KEYS_PER_SEG;

    float ts[TOPK]; int ti[TOPK];
#pragma unroll
    for (int j = 0; j < TOPK; ++j) { ts[j] = -3.0e38f; ti[j] = 0; }

    float* slab = lds[wave];
    v8f z = {0.f,0.f,0.f,0.f,0.f,0.f,0.f,0.f};

    for (int ch = 0; ch < NCHUNKS; ++ch) {
        int key0 = key0seg + ch * CHUNK;
        v8f acc[2][4];
#pragma unroll
        for (int mt = 0; mt < 2; ++mt)
#pragma unroll
            for (int nt = 0; nt < 4; ++nt) acc[mt][nt] = z;

        for (int k0 = 0; k0 < DIM; k0 += 32) {
            v16bf A[2];
#pragma unroll
            for (int mt = 0; mt < 2; ++mt) {
                size_t arow = (size_t)(row0 + mt * 16 + lm);
                const v4u* ap  = (const v4u*)(qb + arow * DIM + k0 + half * 16);
                const v4u* ap2 = (const v4u*)(qb + arow * DIM + k0 + 16 + half * 8);
                A[mt] = frag(ap[0], ap[1]);
            }
#pragma unroll
            for (int nt = 0; nt < 4; ++nt) {
                size_t krow = (size_t)key0 + nt * 16 + lm;
                v16bf B;
                if (KB) {
                    const v4u* bp = (const v4u*)(kb + krow * DIM + k0 + half * 16);
                    B = frag(bp[0], bp[1]);
                } else {
                    const v4f* bp = (const v4f*)(kf + krow * DIM + k0 + half * 16);
                    B = cvt16(bp[0], bp[1], bp[2], bp[3]);
                }
                acc[0][nt] = wmma_bf16(A[0], B, acc[0][nt]);
                acc[1][nt] = wmma_bf16(A[1], B, acc[1][nt]);
            }
        }
        __syncthreads();   // previous scan reads done before overwrite
#pragma unroll
        for (int mt = 0; mt < 2; ++mt)
#pragma unroll
            for (int nt = 0; nt < 4; ++nt)
#pragma unroll
                for (int r = 0; r < 8; ++r)
                    slab[(mt * 16 + r + half * 8) * STRIDE + nt * 16 + lm] = acc[mt][nt][r];
        __syncthreads();

        // each lane owns one of the 32 rows; scan 64 columns, keep private top-16
        float kth = ts[TOPK - 1];
#pragma unroll
        for (int q4 = 0; q4 < 16; ++q4) {
            v4f s4 = *(const v4f*)&slab[lane * STRIDE + q4 * 4];
            float mm = fmaxf(fmaxf(s4[0], s4[1]), fmaxf(s4[2], s4[3]));
            if (mm > kth) {
#pragma unroll
                for (int t = 0; t < 4; ++t) {
                    float s = s4[t];
                    if (s > ts[TOPK - 1])
                        ins16(ts, ti, s, key0 + q4 * 4 + t);
                }
                kth = ts[TOPK - 1];
            }
        }
    }

    // each lane writes its own sorted top-16 partial list
    size_t base = (((size_t)(row0 + lane)) * SEGS + seg) * 16;
#pragma unroll
    for (int j = 0; j < TOPK; ++j) { ps[base + j] = ts[j]; pi[base + j] = ti[j]; }
}

// ---------------- Kernel 4: merge 32 segment-partials per row + softmax ----------------
__global__ void k_merge(const float* __restrict__ ps, const int* __restrict__ pi,
                        float* __restrict__ wout, int* __restrict__ fidx) {
    int row = blockIdx.x * blockDim.x + threadIdx.x;
    if (row >= NBATCH) return;
    float ts[TOPK]; int ti[TOPK];
#pragma unroll
    for (int j = 0; j < TOPK; ++j) { ts[j] = -3.0e38f; ti[j] = 0; }
    for (int seg = 0; seg < SEGS; ++seg) {
        size_t base = ((size_t)row * SEGS + seg) * 16;
        for (int j = 0; j < TOPK; ++j) {
            float s = ps[base + j];
            if (s > ts[TOPK - 1]) ins16(ts, ti, s, pi[base + j]);
            else break; // partial lists are sorted descending
        }
    }
    float mx = ts[0], sum = 0.f, e[TOPK];
#pragma unroll
    for (int j = 0; j < TOPK; ++j) { e[j] = __expf(ts[j] - mx); sum += e[j]; }
    float inv = 1.0f / sum;
#pragma unroll
    for (int j = 0; j < TOPK; ++j) {
        wout[(size_t)row * TOPK + j] = e[j] * inv;
        fidx[(size_t)row * TOPK + j] = ti[j];
    }
}

// ---------------- Kernel 5: gather value rows (one wave per 4KB row) -------------------
__global__ void k_gather(const float* __restrict__ values, const int* __restrict__ fidx,
                         float* __restrict__ out) {
    int wave = threadIdx.x >> 5, lane = threadIdx.x & 31;
    int cp = blockIdx.x * 8 + wave;                 // 0 .. 32767
    int idx = fidx[cp];
    const v4f* src = (const v4f*)(values + (size_t)idx * DIM);
    v4f* dst = (v4f*)(out + (size_t)cp * DIM);
#pragma unroll
    for (int i = 0; i < 8; ++i) dst[lane + i * 32] = src[lane + i * 32];
}

extern "C" void kernel_launch(void* const* d_in, const int* in_sizes, int n_in,
                              void* d_out, int out_size, void* d_ws, size_t ws_size,
                              hipStream_t stream) {
    const float* query  = (const float*)d_in[0];
    const float* keys   = (const float*)d_in[1];
    const float* values = (const float*)d_in[2];
    const float* W      = (const float*)d_in[3];
    const float* bias   = (const float*)d_in[4];

    char* ws = (char*)d_ws;
    bf16_t* qb = (bf16_t*)ws;                               // 4 MB
    float*  ps = (float*)(ws + ((size_t)4 << 20));          // 4 MB
    int*    pi = (int*)  (ws + ((size_t)8 << 20));          // 4 MB
    int*    fidx = (int*)(ws + ((size_t)12 << 20));         // 128 KB
    bf16_t* kb = (bf16_t*)(ws + ((size_t)12 << 20) + ((size_t)256 << 10));
    size_t need_kb = ((size_t)12 << 20) + ((size_t)256 << 10) + (size_t)CAP * DIM * 2;
    bool use_kb = ws_size >= need_kb;

    float* wout = (float*)d_out + (size_t)NBATCH * TOPK * DIM;

    if (use_kb)
        k_cvt_keys<<<(CAP * (DIM / 4)) / 256, 256, 0, stream>>>(keys, kb);
    k_compute_q<<<256, 256, 0, stream>>>(query, W, bias, qb);
    if (use_kb)
        k_scores_topk<true ><<<256, 256, 0, stream>>>(qb, kb, keys, ps, pi);
    else
        k_scores_topk<false><<<256, 256, 0, stream>>>(qb, kb, keys, ps, pi);
    k_merge<<<NBATCH / 256, 256, 0, stream>>>(ps, pi, wout, fidx);
    k_gather<<<(NBATCH * TOPK) / 8, 256, 0, stream>>>(values, fidx, (float*)d_out);
}
